// GraphConvolutionLayer_48576080118434
// MI455X (gfx1250) — matrix operand order repaired and checked by reading, not compiled, not run on hardware
//
#include <hip/hip_runtime.h>
#include <hip/hip_bf16.h>

#define NN 8192
#define FF 128
#define COL_STRIDE 80    // Y slab: 64B K-data + 16B pad per column
#define WCOL_STRIDE 272  // W tile: 256B K-data + 16B pad per column

typedef __attribute__((ext_vector_type(16))) __bf16 v16bf;
typedef __attribute__((ext_vector_type(8)))  float  v8f;
typedef __attribute__((ext_vector_type(4)))  unsigned u32x4;
typedef __attribute__((ext_vector_type(8)))  unsigned u32x8;

union BFrag { v16bf v; uint4 q[2]; };

// ---------------------------------------------------------------------------
// Kernel 1: row sums of A -> d[i] = rsqrt(sum_j A[i,j])
// ---------------------------------------------------------------------------
__global__ __launch_bounds__(256)
void gcn_rowsum(const float* __restrict__ A, float* __restrict__ dvec)
{
    const int row = blockIdx.x;
    const int t   = threadIdx.x;
    const float4* Arow = (const float4*)(A + (size_t)row * NN);
    float s = 0.0f;
#pragma unroll
    for (int i = 0; i < 8; ++i) {
        float4 v = Arow[t + i * 256];
        s += v.x + v.y + v.z + v.w;
    }
    __shared__ float red[256];
    red[t] = s;
    __syncthreads();
    for (int off = 128; off > 0; off >>= 1) {
        if (t < off) red[t] += red[t + off];
        __syncthreads();
    }
    if (t == 0) dvec[row] = rsqrtf(red[0]);
}

// ---------------------------------------------------------------------------
// Kernel 2: Yt[col][j] = bf16( X[j][col] * d[j] )   (K-major for B-fragments)
// ---------------------------------------------------------------------------
__global__ __launch_bounds__(256)
void gcn_build_yt(const float* __restrict__ X, const float* __restrict__ dvec,
                  __bf16* __restrict__ Yt)
{
    int gid = blockIdx.x * 256 + threadIdx.x;      // over NN*FF
    int j = gid >> 7;
    int c = gid & 127;
    Yt[(size_t)c * NN + j] = (__bf16)(X[gid] * dvec[j]);
}

// ---------------------------------------------------------------------------
// Kernel 2b: Wt[o][f] = bf16( W[f][o] )             (K-major for B-fragments)
// ---------------------------------------------------------------------------
__global__ __launch_bounds__(256)
void gcn_build_wt(const float* __restrict__ W, __bf16* __restrict__ Wt)
{
    int gid = blockIdx.x * 256 + threadIdx.x;      // over FF*FF
    int f = gid >> 7;
    int o = gid & 127;
    Wt[o * FF + f] = (__bf16)W[gid];
}

// ---------------------------------------------------------------------------
// TDM: one descriptor DMAs a 32(K) x 128(col) bf16 slab of Yt into LDS.
// D# per ISA ch.8: group0 = count/lds_addr/global_addr/type=2,
// group1 = data_size=2B, pad 16B per 64B (-> 80B LDS column stride),
// tile 32x128, tensor_dim0_stride = NN. Tracked on TENSORcnt.
// ---------------------------------------------------------------------------
__device__ __forceinline__
void tdm_stage(const __bf16* __restrict__ Yt, int k0, unsigned lds_off)
{
    unsigned long long g =
        (unsigned long long)(size_t)Yt + (unsigned long long)k0 * 2u;

    u32x4 g0;
    g0[0] = 1u;                                    // count=1, user mode
    g0[1] = lds_off;                               // lds_addr (bytes)
    g0[2] = (unsigned)(g & 0xffffffffu);           // global_addr[31:0]
    g0[3] = (unsigned)((g >> 32) & 0x01ffffffu)    // global_addr[56:32]
          | (2u << 30);                            // type=2 (image)

    u32x8 g1;
    g1[0] = (1u << 16)                             // data_size = 2B
          | (1u << 20)                             // pad_enable
          | (3u << 22)                             // pad_interval: every 16 DW (64B)
          | (3u << 25);                            // pad_amount: 4 DW (16B)
    g1[1] = ((unsigned)(NN & 0xffff)) << 16;       // tensor_dim0[15:0]
    g1[2] = (unsigned)(NN >> 16) | (128u << 16);   // dim0[31:16] | tensor_dim1[15:0]
    g1[3] = (32u << 16);                           // dim1[31:16]=0 | tile_dim0=32
    g1[4] = 128u;                                  // tile_dim1=128, tile_dim2=0
    g1[5] = (unsigned)NN;                          // tensor_dim0_stride[31:0]
    g1[6] = 0u;
    g1[7] = 0u;

    asm volatile("tensor_load_to_lds %0, %1" :: "s"(g0), "s"(g1) : "memory");
}

// ---------------------------------------------------------------------------
// Kernel 3: agg[i][c] = d[i] * sum_j A[i,j] * Yt[c][j]
// Block = 4 waves = 64 rows. Y K-slab double-buffered in LDS via TDM DMA
// (wave 0 issues, TENSORcnt + barrier sync). A software-pipelined from
// global with on-the-fly fp32->bf16 convert. 8 bf16 WMMAs per 32-K step.
// ---------------------------------------------------------------------------
__global__ __launch_bounds__(128)
void gcn_main_gemm(const float* __restrict__ A, const __bf16* __restrict__ Yt,
                   const float* __restrict__ dvec, float* __restrict__ agg)
{
    __shared__ __align__(16) char ylds[2][128 * COL_STRIDE];

    const int lane    = threadIdx.x & 31;
    const int wave    = threadIdx.x >> 5;
    const int rowbase = blockIdx.x * 64 + wave * 16;
    const int m       = lane & 15;
    const int khalf   = lane >> 4;

    v8f acc[8] = {};
    const float* Arow = A + (size_t)(rowbase + m) * NN;

    const unsigned lbase0 = (unsigned)(size_t)&ylds[0][0];
    const unsigned lbase1 = (unsigned)(size_t)&ylds[1][0];

    // prologue: DMA slab 0 into buffer 0 (one wave drives the TDM)
    if (wave == 0) tdm_stage(Yt, 0, lbase0);

    // prologue: preload raw A registers for k0 = 0
    float4 f0, f1, f2, f3;
    {
        const float4* p0 = (const float4*)(Arow + khalf * 8);
        const float4* p1 = (const float4*)(Arow + 16 + khalf * 8);
        f0 = p0[0]; f1 = p0[1]; f2 = p1[0]; f3 = p1[1];
    }

    for (int k0 = 0; k0 < NN; k0 += 32) {
        const int cur = (k0 >> 5) & 1;

        // wave0 waits its DMA, barrier publishes the slab to all waves and
        // guarantees every wave is done reading the other buffer
        __builtin_amdgcn_s_wait_tensorcnt(0);
        __syncthreads();
        if (wave == 0 && k0 + 32 < NN)
            tdm_stage(Yt, k0 + 32, cur ? lbase0 : lbase1);

        // ---- convert the pipelined A tile to a 16x32 bf16 fragment -------
        v16bf a;
        a[0]  = (__bf16)f0.x; a[1]  = (__bf16)f0.y; a[2]  = (__bf16)f0.z; a[3]  = (__bf16)f0.w;
        a[4]  = (__bf16)f1.x; a[5]  = (__bf16)f1.y; a[6]  = (__bf16)f1.z; a[7]  = (__bf16)f1.w;
        a[8]  = (__bf16)f2.x; a[9]  = (__bf16)f2.y; a[10] = (__bf16)f2.z; a[11] = (__bf16)f2.w;
        a[12] = (__bf16)f3.x; a[13] = (__bf16)f3.y; a[14] = (__bf16)f3.z; a[15] = (__bf16)f3.w;

        // ---- issue next A tile now; consumed next iteration --------------
        const int kn = (k0 + 32 < NN) ? (k0 + 32) : k0;
        {
            const float4* p0 = (const float4*)(Arow + kn + khalf * 8);
            const float4* p1 = (const float4*)(Arow + kn + 16 + khalf * 8);
            f0 = p0[0]; f1 = p0[1]; f2 = p1[0]; f3 = p1[1];
        }
        __builtin_prefetch(Arow + kn + 480, 0, 1);

        // ---- preload all 8 B-fragments, then back-to-back WMMAs ----------
        const char* yb = ylds[cur];
        BFrag b[8];
#pragma unroll
        for (int c = 0; c < 8; ++c) {
            const uint4* bq =
                (const uint4*)(yb + (c * 16 + m) * COL_STRIDE + khalf * 16);
            b[c].q[0] = bq[0];
            b[c].q[1] = bq[2];
        }
#pragma unroll
        for (int c = 0; c < 8; ++c) {
            acc[c] = __builtin_amdgcn_wmma_f32_16x16x32_bf16(
                false, a, false, b[c].v, (short)0, acc[c], false, false);
        }
    }

    // ---- epilogue: scale rows by d[i], store fp32 ------------------------
#pragma unroll
    for (int i = 0; i < 8; ++i) {
        const int row = rowbase + i + khalf * 8;
        const float dd = dvec[row];
#pragma unroll
        for (int c = 0; c < 8; ++c) {
            agg[(size_t)row * FF + c * 16 + m] = acc[c][i] * dd;
        }
    }
}

// ---------------------------------------------------------------------------
// Kernel 4: out = relu(agg @ W + b), K = 128.
// Wt (32KB) staged once per block into LDS via global_load_async_to_lds_b128
// (ASYNCcnt), 272B padded column stride -> conflict-free ds_load_b128.
// ---------------------------------------------------------------------------
__global__ __launch_bounds__(64)
void gcn_out_gemm(const float* __restrict__ agg, const __bf16* __restrict__ Wt,
                  const float* __restrict__ bias, float* __restrict__ out)
{
    __shared__ __align__(16) char wlds[128 * WCOL_STRIDE];

    const int t       = threadIdx.x;
    const int lane    = threadIdx.x & 31;
    const int wave    = threadIdx.x >> 5;
    const int rowbase = (blockIdx.x * 2 + wave) * 16;
    const int m       = lane & 15;
    const int khalf   = lane >> 4;

    // ---- async-stage whole Wt into LDS (32 x b128 per thread) ------------
#pragma unroll
    for (int i = 0; i < 32; ++i) {
        const int idx = i * 64 + t;
        const int col = idx >> 4;      // 0..127
        const int seg = idx & 15;      // 16 x 16B per column
        unsigned long long gaddr =
            (unsigned long long)(size_t)(Wt + col * FF) + seg * 16;
        unsigned loff = (unsigned)(size_t)(wlds + col * WCOL_STRIDE + seg * 16);
        asm volatile("global_load_async_to_lds_b128 %0, %1, off"
                     :: "v"(loff), "v"(gaddr)
                     : "memory");
    }
    asm volatile("s_wait_asynccnt 0x0" ::: "memory");
    __syncthreads();

    v8f acc[8] = {};
    const float* Arow = agg + (size_t)(rowbase + m) * FF;

#pragma unroll
    for (int k0 = 0; k0 < FF; k0 += 32) {
        const float4* p0 = (const float4*)(Arow + k0 + khalf * 8);
        const float4* p1 = (const float4*)(Arow + k0 + 16 + khalf * 8);
        float4 f0 = p0[0], f1 = p0[1];
        float4 f2 = p1[0], f3 = p1[1];
        v16bf a;
        a[0]  = (__bf16)f0.x; a[1]  = (__bf16)f0.y; a[2]  = (__bf16)f0.z; a[3]  = (__bf16)f0.w;
        a[4]  = (__bf16)f1.x; a[5]  = (__bf16)f1.y; a[6]  = (__bf16)f1.z; a[7]  = (__bf16)f1.w;
        a[8]  = (__bf16)f2.x; a[9]  = (__bf16)f2.y; a[10] = (__bf16)f2.z; a[11] = (__bf16)f2.w;
        a[12] = (__bf16)f3.x; a[13] = (__bf16)f3.y; a[14] = (__bf16)f3.z; a[15] = (__bf16)f3.w;

        BFrag b[8];
#pragma unroll
        for (int c = 0; c < 8; ++c) {
            const uint4* bq =
                (const uint4*)(wlds + (c * 16 + m) * WCOL_STRIDE + k0 * 2 + khalf * 16);
            b[c].q[0] = bq[0];
            b[c].q[1] = bq[2];
        }
#pragma unroll
        for (int c = 0; c < 8; ++c) {
            acc[c] = __builtin_amdgcn_wmma_f32_16x16x32_bf16(
                false, a, false, b[c].v, (short)0, acc[c], false, false);
        }
    }

#pragma unroll
    for (int c = 0; c < 8; ++c) {
        const float bb = bias[c * 16 + m];
#pragma unroll
        for (int i = 0; i < 8; ++i) {
            const int row = rowbase + i + khalf * 8;
            float v = acc[c][i] + bb;
            out[(size_t)row * FF + c * 16 + m] = fmaxf(v, 0.0f);
        }
    }
}

// ---------------------------------------------------------------------------
// Host-side launch
// ---------------------------------------------------------------------------
extern "C" void kernel_launch(void* const* d_in, const int* in_sizes, int n_in,
                              void* d_out, int out_size, void* d_ws, size_t ws_size,
                              hipStream_t stream)
{
    const float* A = (const float*)d_in[0];   // [8192, 8192]
    const float* X = (const float*)d_in[1];   // [8192, 128]
    const float* W = (const float*)d_in[2];   // [128, 128]
    const float* b = (const float*)d_in[3];   // [128]
    float* out = (float*)d_out;               // [8192, 128]

    // Workspace layout (bytes):
    //   d    : [NN] f32            @ 0          (32 KB)
    //   Yt   : [FF][NN] bf16       @ 32768      (2 MB)
    //   Wt   : [FF][FF] bf16       @ 2129920    (32 KB)
    //   agg  : [NN][FF] f32        @ 2162688    (4 MB)
    char* ws = (char*)d_ws;
    float*  dvec = (float*)(ws);
    __bf16* Yt   = (__bf16*)(ws + 32768);
    __bf16* Wt   = (__bf16*)(ws + 2129920);
    float*  agg  = (float*)(ws + 2162688);

    gcn_rowsum<<<NN, 256, 0, stream>>>(A, dvec);
    gcn_build_yt<<<(NN * FF) / 256, 256, 0, stream>>>(X, dvec, Yt);
    gcn_build_wt<<<(FF * FF) / 256, 256, 0, stream>>>(W, Wt);
    gcn_main_gemm<<<NN / 64, 128, 0, stream>>>(A, Yt, dvec, agg);
    gcn_out_gemm<<<NN / 32, 64, 0, stream>>>(agg, Wt, b, out);
}
